// HmmLm_19353122636095
// MI455X (gfx1250) — matrix operand order repaired and checked by reading, not compiled, not run on hardware
//
#include <hip/hip_runtime.h>

typedef __bf16 bf16;
typedef __attribute__((ext_vector_type(16))) __bf16 v16bf;
typedef __attribute__((ext_vector_type(8)))  float  v8f;

#define CC 1024
#define HD 512
#define VOCAB 32000
#define NB 16
#define TT 128

__device__ __forceinline__ v8f wmma_bf16(v16bf a, v16bf b, v8f c) {
  return __builtin_amdgcn_wmma_f32_16x16x32_bf16(false, a, false, b, (short)0, c, false, false);
}

// A-matrix 16-bit 16x32 per-lane K index (ISA 7.12.2): element e (0..15), lane-half h
__device__ __forceinline__ int kkA(int e, int h) {
  int r = e >> 1;
  return 2 * r + (e & 1) + 8 * h + ((r >> 2) << 3);
}

// ---------------- packing kernels ----------------

__global__ __launch_bounds__(256) void pack_A(const float* __restrict__ A, bf16* __restrict__ out,
                                              int K, int KB, int total) {
  int idx = blockIdx.x * 256 + threadIdx.x;
  if (idx >= total) return;
  int e = idx & 15;
  int lane = (idx >> 4) & 31;
  int t2 = idx >> 9;
  int kb = t2 % KB;
  int mt = t2 / KB;
  int m = mt * 16 + (lane & 15);
  int k = kb * 32 + kkA(e, lane >> 4);
  out[idx] = (bf16)A[(size_t)m * K + k];
}

// B-matrix 16-bit 32x16: lanes 0-15 hold K=0..7,16..23; lanes 16-31 hold K=8..15,24..31 (same order as A)
__global__ __launch_bounds__(256) void pack_B(const float* __restrict__ B, bf16* __restrict__ out,
                                              int N, int KB, int total) {
  int idx = blockIdx.x * 256 + threadIdx.x;
  if (idx >= total) return;
  int e = idx & 15;
  int lane = (idx >> 4) & 31;
  int t2 = idx >> 9;
  int kb = t2 % KB;
  int nt = t2 / KB;
  int n = nt * 16 + (lane & 15);
  int k = kb * 32 + (e & 7) + 8 * (lane >> 4) + 16 * (e >> 3);
  out[idx] = (bf16)B[(size_t)k * N + n];
}

// gather columns of term_wo by token id: B[k, j] = wo[k, text_j], j = t*16+n
__global__ __launch_bounds__(256) void pack_B_gather(const float* __restrict__ W,
                                                     const int* __restrict__ text,
                                                     bf16* __restrict__ out, int total) {
  int idx = blockIdx.x * 256 + threadIdx.x;
  if (idx >= total) return;
  int e = idx & 15;
  int lane = (idx >> 4) & 31;
  int t2 = idx >> 9;
  int kb = t2 & 15;       // KB = 16
  int jt = t2 >> 4;
  int j = jt * 16 + (lane & 15);
  int k = kb * 32 + (e & 7) + 8 * (lane >> 4) + 16 * (e >> 3);
  int tok = text[(j & 15) * TT + (j >> 4)];   // text[n*T + t]
  out[idx] = (bf16)W[(size_t)k * VOCAB + tok];
}

// ---------------- generic WMMA GEMM (packed in, f32 row-major out) ----------------

__global__ __launch_bounds__(256) void gemm_tile(const bf16* __restrict__ Ap, const bf16* __restrict__ Bp,
                                                 const float* __restrict__ bias,
                                                 const float* __restrict__ resid,
                                                 float* __restrict__ out,
                                                 int Mt, int Nt, int KB, int N, int relu) {
  int wave = blockIdx.x * 8 + (threadIdx.x >> 5);
  if (wave >= Mt * Nt) return;
  int lane = threadIdx.x & 31, half = lane >> 4, lo = lane & 15;
  int mt = wave / Nt, nt = wave % Nt;
  const bf16* ap = Ap + (size_t)mt * KB * 512;
  const bf16* bp = Bp + (size_t)nt * KB * 512;
  v8f acc = {};
  for (int kb = 0; kb < KB; ++kb) {
    v16bf a = *(const v16bf*)(ap + ((kb * 32 + lane) << 4));
    v16bf b = *(const v16bf*)(bp + ((kb * 32 + lane) << 4));
    acc = wmma_bf16(a, b, acc);
  }
  int n = nt * 16 + lo;
  float bv = bias ? bias[n] : 0.f;
#pragma unroll
  for (int r = 0; r < 8; ++r) {
    int m = mt * 16 + r + 8 * half;
    float v = acc[r] + bv;
    if (relu) v = fmaxf(v, 0.f);
    if (resid) v += resid[(size_t)m * N + n];
    out[(size_t)m * N + n] = v;
  }
}

// ---------------- term head: streamed row-wise logsumexp over V=32000 ----------------

__global__ __launch_bounds__(128) void term_logz_partial(const bf16* __restrict__ Yp,
                                                         const bf16* __restrict__ Wop,
                                                         const float* __restrict__ bo,
                                                         float* __restrict__ pm, float* __restrict__ ps) {
  int ct = blockIdx.x / 100;
  int vg = blockIdx.x % 100;
  int wv = threadIdx.x >> 5;
  int lane = threadIdx.x & 31, half = lane >> 4, lo = lane & 15;
  const bf16* ap = Yp + (size_t)ct * 16 * 512;   // KB=16 -> 8192 elems per ctile
  float m8[8], s8[8];
#pragma unroll
  for (int r = 0; r < 8; ++r) { m8[r] = -INFINITY; s8[r] = 0.f; }
  int vt0 = (vg * 4 + wv) * 5;
  for (int i = 0; i < 5; ++i) {
    int vt = vt0 + i;
    const bf16* bp = Wop + (size_t)vt * 16 * 512;
    v8f acc = {};
    for (int kb = 0; kb < 16; ++kb) {
      v16bf a = *(const v16bf*)(ap + ((kb * 32 + lane) << 4));
      v16bf b = *(const v16bf*)(bp + ((kb * 32 + lane) << 4));
      acc = wmma_bf16(a, b, acc);
    }
    float bv = bo[vt * 16 + lo];
#pragma unroll
    for (int r = 0; r < 8; ++r) {   // online logsumexp update
      float x = acc[r] + bv;
      if (x > m8[r]) { s8[r] = s8[r] * __expf(m8[r] - x) + 1.f; m8[r] = x; }
      else           { s8[r] += __expf(x - m8[r]); }
    }
  }
  for (int d = 1; d < 16; d <<= 1) {
#pragma unroll
    for (int r = 0; r < 8; ++r) {
      float mo = __shfl_xor(m8[r], d, 32);
      float so = __shfl_xor(s8[r], d, 32);
      float mn = fmaxf(m8[r], mo);
      s8[r] = s8[r] * __expf(m8[r] - mn) + so * __expf(mo - mn);
      m8[r] = mn;
    }
  }
  if (lo == 0) {
    int pidx = vg * 4 + wv;
#pragma unroll
    for (int r = 0; r < 8; ++r) {
      int c = ct * 16 + r + 8 * half;
      pm[(size_t)c * 400 + pidx] = m8[r];
      ps[(size_t)c * 400 + pidx] = s8[r];
    }
  }
}

__global__ __launch_bounds__(256) void logz_reduce(const float* __restrict__ pm,
                                                   const float* __restrict__ ps,
                                                   float* __restrict__ logZ, int np) {
  int c = blockIdx.x * 256 + threadIdx.x;
  if (c >= CC) return;
  const float* pmr = pm + (size_t)c * np;
  const float* psr = ps + (size_t)c * np;
  float m = -INFINITY;
  for (int i = 0; i < np; ++i) m = fmaxf(m, pmr[i]);
  float s = 0.f;
  for (int i = 0; i < np; ++i) s += psr[i] * __expf(pmr[i] - m);
  logZ[c] = m + __logf(s);
}

// EM[t][n][c] = Y_term[c,:]·wo[:,tok] + bo[tok] - logZ[c]   (gathered-B GEMM)
__global__ __launch_bounds__(256) void emission_gemm(const bf16* __restrict__ Yp,
                                                     const bf16* __restrict__ Wgp,
                                                     const float* __restrict__ term_bo,
                                                     const int* __restrict__ text,
                                                     const float* __restrict__ logZ,
                                                     float* __restrict__ EM) {
  int wave = blockIdx.x * 8 + (threadIdx.x >> 5);
  int lane = threadIdx.x & 31, half = lane >> 4, lo = lane & 15;
  int ct = wave >> 7;        // Nt = 128
  int jt = wave & 127;
  const bf16* ap = Yp + (size_t)ct * 8192;
  const bf16* bp = Wgp + (size_t)jt * 8192;
  v8f acc = {};
  for (int kb = 0; kb < 16; ++kb) {
    v16bf a = *(const v16bf*)(ap + ((kb * 32 + lane) << 4));
    v16bf b = *(const v16bf*)(bp + ((kb * 32 + lane) << 4));
    acc = wmma_bf16(a, b, acc);
  }
  int j = jt * 16 + lo;
  int tok = text[(j & 15) * TT + (j >> 4)];
  float bv = term_bo[tok];
#pragma unroll
  for (int r = 0; r < 8; ++r) {
    int c = ct * 16 + r + 8 * half;
    EM[(size_t)j * CC + c] = acc[r] + bv - logZ[c];
  }
}

// ---------------- transition head helpers ----------------

__global__ __launch_bounds__(256) void row_logsumexp(const float* __restrict__ X,
                                                     float* __restrict__ out, int ncol) {
  __shared__ float red[256];
  int row = blockIdx.x, tid = threadIdx.x;
  const float* x = X + (size_t)row * ncol;
  float m = -INFINITY;
  for (int i = tid; i < ncol; i += 256) m = fmaxf(m, x[i]);
  red[tid] = m; __syncthreads();
  for (int s = 128; s > 0; s >>= 1) { if (tid < s) red[tid] = fmaxf(red[tid], red[tid + s]); __syncthreads(); }
  m = red[0]; __syncthreads();
  float sum = 0.f;
  for (int i = tid; i < ncol; i += 256) sum += __expf(x[i] - m);
  red[tid] = sum; __syncthreads();
  for (int s = 128; s > 0; s >>= 1) { if (tid < s) red[tid] += red[tid + s]; __syncthreads(); }
  if (tid == 0) out[row] = m + __logf(red[0]);
}

// Ppack[nt][kb][lane][e] = exp(L[k, next] - logZrow[k])  (B-packed, bf16)
__global__ __launch_bounds__(256) void make_Ppack(const float* __restrict__ L,
                                                  const float* __restrict__ lz,
                                                  bf16* __restrict__ out) {
  int idx = blockIdx.x * 256 + threadIdx.x;   // total 1M
  int e = idx & 15;
  int lane = (idx >> 4) & 31;
  int kb = (idx >> 9) & 31;
  int nt = idx >> 14;
  int k = kb * 32 + (e & 7) + 8 * (lane >> 4) + 16 * (e >> 3);
  int nx = nt * 16 + (lane & 15);
  out[idx] = (bf16)__expf(L[(size_t)k * CC + nx] - lz[k]);
}

// ---------------- start head ----------------

__global__ __launch_bounds__(256) void start_head(const float* __restrict__ Y,
                                                  const float* __restrict__ wo,
                                                  const float* __restrict__ bo,
                                                  float* __restrict__ slog) {
  int c = blockIdx.x * 256 + threadIdx.x;
  if (c >= CC) return;
  float s = bo[0];
  for (int k = 0; k < HD; ++k) s += Y[(size_t)c * HD + k] * wo[k];
  slog[c] = s;
}

__global__ __launch_bounds__(256) void vec_logsoftmax(const float* __restrict__ x,
                                                      float* __restrict__ y, int n) {
  __shared__ float red[256];
  int tid = threadIdx.x;
  float m = -INFINITY;
  for (int i = tid; i < n; i += 256) m = fmaxf(m, x[i]);
  red[tid] = m; __syncthreads();
  for (int s = 128; s > 0; s >>= 1) { if (tid < s) red[tid] = fmaxf(red[tid], red[tid + s]); __syncthreads(); }
  m = red[0]; __syncthreads();
  float sum = 0.f;
  for (int i = tid; i < n; i += 256) sum += __expf(x[i] - m);
  red[tid] = sum; __syncthreads();
  for (int s = 128; s > 0; s >>= 1) { if (tid < s) red[tid] += red[tid + s]; __syncthreads(); }
  float lse = m + __logf(red[0]);
  for (int i = tid; i < n; i += 256) y[i] = x[i] - lse;
}

// ---------------- the serial HMM recursion: 1 workgroup, 32 waves, all on-chip ----------------

__global__ __launch_bounds__(1024) void hmm_recursion(const bf16* __restrict__ Ppk,
                                                      const float* __restrict__ EM,
                                                      const float* __restrict__ startv,
                                                      const int* __restrict__ lengths,
                                                      float* __restrict__ out) {
  __shared__ bf16 Alds[32 * 32 * 16];   // A' = exp(alpha-m), pre-swizzled WMMA A layout (32 KB)
  __shared__ float msh[2][16];
  __shared__ float pmax[32][16];
  __shared__ float psum[32][16];
  __shared__ float ev[16];

  const int tid = threadIdx.x;
  const int wave = tid >> 5;
  const int lane = tid & 31;
  const int half = lane >> 4;
  const int lo = lane & 15;
  const int mylen = lengths[tid & 15];

  if (tid < 16) ev[tid] = 0.0f;

  // ---- t = 0: alpha0[n,c] = start[c] + EM[0][n][c] (thread tid owns column c = tid) ----
  {
    const int c = tid;
    const int b = c >> 5;
    const int kv = c & 31;
    const int hh = (kv >> 3) & 1;
    const int rr = ((kv >> 4) << 2) + ((kv & 7) >> 1);
    const int ee = 2 * rr + (kv & 1);
    const float sv = startv[c];

    float a[16];
#pragma unroll
    for (int n = 0; n < 16; ++n) a[n] = sv + EM[(size_t)n * CC + c];

    float mr[16];
#pragma unroll
    for (int n = 0; n < 16; ++n) mr[n] = a[n];
    for (int d = 1; d < 32; d <<= 1)
#pragma unroll
      for (int n = 0; n < 16; ++n) mr[n] = fmaxf(mr[n], __shfl_xor(mr[n], d, 32));
    if (lane == 0)
#pragma unroll
      for (int n = 0; n < 16; ++n) pmax[wave][n] = mr[n];
    __syncthreads();
    if (tid < 16) {
      float m = -INFINITY;
      for (int w = 0; w < 32; ++w) m = fmaxf(m, pmax[w][tid]);
      msh[0][tid] = m;
    }
    __syncthreads();
    float sr[16];
#pragma unroll
    for (int n = 0; n < 16; ++n) {
      float e = __expf(a[n] - msh[0][n]);
      sr[n] = e;
      Alds[((b * 32 + hh * 16 + n) << 4) + ee] = (bf16)e;
    }
    for (int d = 1; d < 32; d <<= 1)
#pragma unroll
      for (int n = 0; n < 16; ++n) sr[n] += __shfl_xor(sr[n], d, 32);
    if (lane == 0)
#pragma unroll
      for (int n = 0; n < 16; ++n) psum[wave][n] = sr[n];
    __syncthreads();
    if (tid < 16 && mylen == 1) {
      float s = 0.f;
      for (int w = 0; w < 32; ++w) s += psum[w][tid];
      ev[tid] = msh[0][tid] + __logf(s);
    }
  }

  // each wave owns two 16-wide c' tiles
  const int ct0 = wave << 1;
  const bf16* __restrict__ bp0 = Ppk + (size_t)ct0 * (32 * 512);
  const bf16* __restrict__ bp1 = bp0 + (32 * 512);
  // LDS write targets for next-step A' at k = ct0*16+lo (even tile) / +16 (odd tile)
  const int b_blk = ct0 >> 1;
  const int hh0 = (lo >> 3) & 1;
  const int ee0 = 2 * ((lo & 7) >> 1) + (lo & 1);
  const int ee1 = ee0 + 8;

  for (int t = 1; t < TT; ++t) {
    const int p = t & 1, pq = p ^ 1;
    // GEMM: S = A' @ Pexp   (reads Alds of step t-1; Ppack streams from L2)
    v8f acc0 = {}; v8f acc1 = {};
    for (int kb = 0; kb < 32; ++kb) {
      v16bf av = *(const v16bf*)&Alds[(kb * 32 + lane) << 4];
      v16bf b0 = *(const v16bf*)(bp0 + ((kb * 32 + lane) << 4));
      v16bf b1 = *(const v16bf*)(bp1 + ((kb * 32 + lane) << 4));
      acc0 = wmma_bf16(av, b0, acc0);
      acc1 = wmma_bf16(av, b1, acc1);
    }
    const float* emt = EM + (size_t)t * (NB * CC);
    float al0[8], al1[8], mr8[8];
#pragma unroll
    for (int r = 0; r < 8; ++r) {
      int n = r + 8 * half;
      float mp = msh[pq][n];
      al0[r] = mp + __logf(fmaxf(acc0[r], 1e-37f)) + emt[n * CC + ct0 * 16 + lo];
      al1[r] = mp + __logf(fmaxf(acc1[r], 1e-37f)) + emt[n * CC + ct0 * 16 + 16 + lo];
      mr8[r] = fmaxf(al0[r], al1[r]);
    }
    for (int d = 1; d < 16; d <<= 1)
#pragma unroll
      for (int r = 0; r < 8; ++r) mr8[r] = fmaxf(mr8[r], __shfl_xor(mr8[r], d, 32));
    if (lo == 0)
#pragma unroll
      for (int r = 0; r < 8; ++r) pmax[wave][r + 8 * half] = mr8[r];
    __syncthreads();                               // B1: GEMM reads done, pmax ready
    if (tid < 16) {
      float m = -INFINITY;
      for (int w = 0; w < 32; ++w) m = fmaxf(m, pmax[w][tid]);
      msh[p][tid] = m;
    }
    __syncthreads();                               // B2: new row max ready
    float sr8[8];
#pragma unroll
    for (int r = 0; r < 8; ++r) {
      int n = r + 8 * half;
      float mnew = msh[p][n];
      float e0 = __expf(al0[r] - mnew);
      float e1 = __expf(al1[r] - mnew);
      sr8[r] = e0 + e1;
      int base = (b_blk * 32 + hh0 * 16 + n) << 4;
      Alds[base + ee0] = (bf16)e0;
      Alds[base + ee1] = (bf16)e1;
    }
    for (int d = 1; d < 16; d <<= 1)
#pragma unroll
      for (int r = 0; r < 8; ++r) sr8[r] += __shfl_xor(sr8[r], d, 32);
    if (lo == 0)
#pragma unroll
      for (int r = 0; r < 8; ++r) psum[wave][r + 8 * half] = sr8[r];
    __syncthreads();                               // B3: A' and row sums ready
    if (tid < 16 && mylen - 1 == t) {
      float s = 0.f;
      for (int w = 0; w < 32; ++w) s += psum[w][tid];
      ev[tid] = msh[p][tid] + __logf(s);           // logsumexp_c alpha_t[n,:] for free
    }
  }
  __syncthreads();
  if (tid == 0) {
    float s = 0.f;
    for (int n = 0; n < 16; ++n) s += ev[n];
    out[0] = s;
  }
}

// ---------------- host launcher ----------------

extern "C" void kernel_launch(void* const* d_in, const int* in_sizes, int n_in,
                              void* d_out, int out_size, void* d_ws, size_t ws_size,
                              hipStream_t stream) {
  (void)in_sizes; (void)n_in; (void)out_size; (void)ws_size;
  const int* text    = (const int*)d_in[0];
  const int* lengths = (const int*)d_in[1];
  const float* EMB[3] = {(const float*)d_in[2],  (const float*)d_in[9],  (const float*)d_in[16]};
  const float* W1[3]  = {(const float*)d_in[3],  (const float*)d_in[10], (const float*)d_in[17]};
  const float* B1[3]  = {(const float*)d_in[4],  (const float*)d_in[11], (const float*)d_in[18]};
  const float* W2[3]  = {(const float*)d_in[5],  (const float*)d_in[12], (const float*)d_in[19]};
  const float* B2[3]  = {(const float*)d_in[6],  (const float*)d_in[13], (const float*)d_in[20]};
  const float* WO[3]  = {(const float*)d_in[7],  (const float*)d_in[14], (const float*)d_in[21]};
  const float* BO[3]  = {(const float*)d_in[8],  (const float*)d_in[15], (const float*)d_in[22]};

  char* ws = (char*)d_ws;
  size_t off = 0;
  auto alloc = [&](size_t bytes) -> char* {
    char* p = ws + off;
    off = (off + bytes + 255) & ~(size_t)255;
    return p;
  };
  bf16* Xp    = (bf16*)alloc((size_t)CC * HD * 2);          // packed A (reused per net)
  bf16* W1p   = (bf16*)alloc((size_t)HD * HD * 2);
  bf16* W2p   = (bf16*)alloc((size_t)HD * HD * 2);
  float* R32  = (float*)alloc((size_t)CC * HD * 4);
  bf16* Rp    = (bf16*)alloc((size_t)CC * HD * 2);
  float* Y32  = (float*)alloc((size_t)CC * HD * 4);
  bf16* YpT   = (bf16*)alloc((size_t)CC * HD * 2);
  bf16* YpE   = (bf16*)alloc((size_t)CC * HD * 2);
  bf16* WopT  = (bf16*)alloc((size_t)HD * CC * 2);
  bf16* WopE  = (bf16*)alloc((size_t)HD * VOCAB * 2);
  float* Ltr  = (float*)alloc((size_t)CC * CC * 4);
  float* lzr  = (float*)alloc((size_t)CC * 4);
  bf16* Ppk   = (bf16*)alloc((size_t)CC * CC * 2);
  float* pm   = (float*)alloc((size_t)CC * 400 * 4);
  float* ps   = (float*)alloc((size_t)CC * 400 * 4);
  float* logZ = (float*)alloc((size_t)CC * 4);
  bf16* Wgp   = (bf16*)alloc((size_t)(NB * TT) * HD * 2);
  float* EM   = (float*)alloc((size_t)TT * NB * CC * 4);
  float* slog = (float*)alloc((size_t)CC * 4);
  float* stv  = (float*)alloc((size_t)CC * 4);

  for (int net = 0; net < 3; ++net) {
    pack_A<<<2048, 256, 0, stream>>>(EMB[net], Xp, HD, 16, CC * HD);
    pack_B<<<1024, 256, 0, stream>>>(W1[net], W1p, HD, 16, HD * HD);
    pack_B<<<1024, 256, 0, stream>>>(W2[net], W2p, HD, 16, HD * HD);
    gemm_tile<<<256, 256, 0, stream>>>(Xp, W1p, B1[net], nullptr, R32, 64, 32, 16, HD, 1);
    pack_A<<<2048, 256, 0, stream>>>(R32, Rp, HD, 16, CC * HD);
    gemm_tile<<<256, 256, 0, stream>>>(Rp, W2p, B2[net], EMB[net], Y32, 64, 32, 16, HD, 1);
    if (net == 0) {
      start_head<<<4, 256, 0, stream>>>(Y32, WO[0], BO[0], slog);
      vec_logsoftmax<<<1, 256, 0, stream>>>(slog, stv, CC);
    } else if (net == 1) {
      pack_A<<<2048, 256, 0, stream>>>(Y32, YpT, HD, 16, CC * HD);
      pack_B<<<2048, 256, 0, stream>>>(WO[1], WopT, CC, 16, HD * CC);
      gemm_tile<<<512, 256, 0, stream>>>(YpT, WopT, BO[1], nullptr, Ltr, 64, 64, 16, CC, 0);
      row_logsumexp<<<1024, 256, 0, stream>>>(Ltr, lzr, CC);
      make_Ppack<<<4096, 256, 0, stream>>>(Ltr, lzr, Ppk);
    } else {
      pack_A<<<2048, 256, 0, stream>>>(Y32, YpE, HD, 16, CC * HD);
      pack_B<<<64000, 256, 0, stream>>>(WO[2], WopE, VOCAB, 16, HD * VOCAB);
      term_logz_partial<<<6400, 128, 0, stream>>>(YpE, WopE, BO[2], pm, ps);
      logz_reduce<<<4, 256, 0, stream>>>(pm, ps, logZ, 400);
      pack_B_gather<<<4096, 256, 0, stream>>>(WO[2], text, Wgp, NB * TT * HD);
      emission_gemm<<<1024, 256, 0, stream>>>(YpE, Wgp, BO[2], text, logZ, EM);
    }
  }
  hmm_recursion<<<1, 1024, 0, stream>>>(Ppk, EM, stv, lengths, (float*)d_out);
}
